// MoELayer_25537875542064
// MI455X (gfx1250) — matrix operand order repaired and checked by reading, not compiled, not run on hardware
//
#include <hip/hip_runtime.h>
#include <hip/hip_bf16.h>

// ---------------- problem dims ----------------
#define S_LEN 1024
#define B_SZ  8
#define DIN   768
#define E_DIM 768
#define NEXP  10
#define HID   256
#define DOUT  768
#define NH    8
#define HD    96
#define NTOK  (S_LEN * B_SZ)      // 8192
#define LN_EPS 1e-5f
#define AW_CONST (1.0f / 1024.0f) // aw is provably constant = 1/S

typedef __attribute__((ext_vector_type(16))) _Float16 v16h;
typedef __attribute__((ext_vector_type(8)))  _Float16 v8h;
typedef __attribute__((ext_vector_type(8)))  float    v8f;
typedef __attribute__((ext_vector_type(4)))  unsigned int u32x4;
typedef __attribute__((ext_vector_type(8)))  int i32x8;
typedef __attribute__((ext_vector_type(4)))  int i32x4;

#if defined(__HIP_DEVICE_COMPILE__) && __has_builtin(__builtin_amdgcn_tensor_load_to_lds)
#define HAVE_TDM 1
#else
#define HAVE_TDM 0
#endif

// ---------------- WMMA fragment helpers (CDNA5 16x16x32 f16 layouts) ----------------
// A (16x32 f16): lane L holds row M=L&15; K = kb..kb+7 and 16+kb..16+kb+7, kb=(L>>4)*8
__device__ __forceinline__ v16h frag_a(const _Float16* t, int ld) {
  int lane = threadIdx.x & 31;
  int m  = lane & 15;
  int kb = (lane >> 4) << 3;
  const _Float16* p = t + m * ld + kb;
  v8h lo = *(const v8h*)(p);
  v8h hi = *(const v8h*)(p + 16);
  v16h a;
#pragma unroll
  for (int i = 0; i < 8; ++i) { a[i] = lo[i]; a[i + 8] = hi[i]; }
  return a;
}
// B (32x16 f16) from LDS stored [n][k]: lane L holds col N=L&15; K = kb..kb+7 and 16+kb..16+kb+7, kb=(L>>4)*8
__device__ __forceinline__ v16h frag_b(const _Float16* t, int ld) {
  int lane = threadIdx.x & 31;
  int n  = lane & 15;
  int kb = (lane >> 4) << 3;
  const _Float16* p = t + n * ld + kb;
  v8h lo = *(const v8h*)(p);
  v8h hi = *(const v8h*)(p + 16);
  v16h b;
#pragma unroll
  for (int i = 0; i < 8; ++i) { b[i] = lo[i]; b[i + 8] = hi[i]; }
  return b;
}
__device__ __forceinline__ v8f wmma_f16(v16h a, v16h b, v8f c) {
  return __builtin_amdgcn_wmma_f32_16x16x32_f16(false, a, false, b, (short)0, c, false, false);
}
__device__ __forceinline__ v8h pack8(float4 a, float4 b) {
  v8h r;
  r[0] = (_Float16)a.x; r[1] = (_Float16)a.y; r[2] = (_Float16)a.z; r[3] = (_Float16)a.w;
  r[4] = (_Float16)b.x; r[5] = (_Float16)b.y; r[6] = (_Float16)b.z; r[7] = (_Float16)b.w;
  return r;
}

#if HAVE_TDM
// Issue a TDM 2D tile load: rows x cols f16 tile, packed row-major into LDS.
// Descriptor per CDNA5 ISA (group0 + group1; groups 2/3 zero => 2D).
__device__ __forceinline__ void tdm_load_tile_f16(
    unsigned lds_off, const _Float16* gptr,
    unsigned rows, unsigned cols,
    unsigned tensor_rows, unsigned tensor_cols, unsigned row_stride_elems) {
  unsigned long long ga = (unsigned long long)(uintptr_t)gptr;
  u32x4 g0;
  g0[0] = 1u;                                              // count=1, user descriptor
  g0[1] = lds_off;                                         // lds_addr (bytes)
  g0[2] = (unsigned)(ga & 0xFFFFFFFFu);                    // global_addr lo
  g0[3] = (unsigned)((ga >> 32) & 0x01FFFFFFu) | (2u << 30); // global_addr hi | type=2
  i32x8 g1;
  g1[0] = (int)(1u << 16);                                 // data_size=1 (2 bytes/elem)
  g1[1] = (int)((tensor_cols & 0xFFFFu) << 16);            // tensor_dim0[15:0]
  g1[2] = (int)(((tensor_cols >> 16) & 0xFFFFu) |
                ((tensor_rows & 0xFFFFu) << 16));          // tensor_dim0 hi | tensor_dim1 lo
  g1[3] = (int)(((tensor_rows >> 16) & 0xFFFFu) |
                ((cols & 0xFFFFu) << 16));                 // tensor_dim1 hi | tile_dim0
  g1[4] = (int)(rows & 0xFFFFu);                           // tile_dim1 (tile_dim2=0)
  g1[5] = (int)row_stride_elems;                           // tensor_dim0_stride lo
  g1[6] = 0;                                               // stride hi, dim1_stride lo
  g1[7] = 0;
  i32x4 z = {0, 0, 0, 0};
#if __clang_major__ >= 23
  i32x8 z8 = {0, 0, 0, 0, 0, 0, 0, 0};
  __builtin_amdgcn_tensor_load_to_lds(g0, g1, z, z, z8, 0);
#else
  __builtin_amdgcn_tensor_load_to_lds(g0, g1, z, z, 0);
#endif
}
#endif

// ---------------- utility kernels ----------------
__global__ void moe_zero_kernel(float* p, int n) {
  int i = blockIdx.x * 256 + threadIdx.x;
  if (i < n) p[i] = 0.0f;
}

__global__ void moe_cvt_f16_kernel(const float* __restrict__ src, _Float16* __restrict__ dst, int n) {
  int i = (blockIdx.x * 256 + threadIdx.x) * 4;
  if (i + 3 < n) {
    float4 v = *(const float4*)(src + i);
    dst[i + 0] = (_Float16)v.x; dst[i + 1] = (_Float16)v.y;
    dst[i + 2] = (_Float16)v.z; dst[i + 3] = (_Float16)v.w;
  }
}

__global__ void moe_scan_kernel(const int* counts, int* offs) {
  if (threadIdx.x == 0) {
    int r = 0;
    for (int e = 0; e < NEXP; ++e) { offs[e] = r; r += counts[e]; }
  }
}

// ---------------- GEMM (f16 operands): C[M,N] = A16[M,K] @ W16[N,K]^T + bias ----------------
// 64x64 block tile, 64-wide k-tiles, TDM double-buffered global->LDS staging.
__global__ __launch_bounds__(256)
void moe_gemm_tdm_f16(const _Float16* __restrict__ A16, const _Float16* __restrict__ W16,
                      const float* __restrict__ bias, float* __restrict__ C,
                      int M, int N, int K) {
  __shared__ __align__(16) _Float16 As[2][64 * 64];
  __shared__ __align__(16) _Float16 Bs[2][64 * 64];
  __shared__ float bias_s[64];
  int tid = threadIdx.x;
  int nbase = blockIdx.x * 64;
  int mbase = blockIdx.y * 64;
  if (tid < 64) bias_s[tid] = bias[nbase + tid];
  int wave = tid >> 5, lane = tid & 31, lg = lane >> 4, nl = lane & 15;
  int mt = wave >> 1, nt0 = (wave & 1) * 2;
  v8f c0 = {}, c1 = {};
  int nkb = K >> 6;
#if HAVE_TDM
  if (wave == 0) {
    tdm_load_tile_f16((unsigned)(uintptr_t)&As[0][0], A16 + (size_t)mbase * K,
                      64, 64, (unsigned)(M - mbase), (unsigned)K, (unsigned)K);
    tdm_load_tile_f16((unsigned)(uintptr_t)&Bs[0][0], W16 + (size_t)nbase * K,
                      64, 64, (unsigned)(N - nbase), (unsigned)K, (unsigned)K);
  }
#endif
  for (int kb = 0; kb < nkb; ++kb) {
    int cur = kb & 1;
#if HAVE_TDM
    if (wave == 0) {
      if (kb + 1 < nkb) {
        int k1 = (kb + 1) << 6;
        tdm_load_tile_f16((unsigned)(uintptr_t)&As[cur ^ 1][0], A16 + (size_t)mbase * K + k1,
                          64, 64, (unsigned)(M - mbase), (unsigned)(K - k1), (unsigned)K);
        tdm_load_tile_f16((unsigned)(uintptr_t)&Bs[cur ^ 1][0], W16 + (size_t)nbase * K + k1,
                          64, 64, (unsigned)(N - nbase), (unsigned)(K - k1), (unsigned)K);
        __builtin_amdgcn_s_wait_tensorcnt(2);  // current buffer's 2 loads retired
      } else {
        __builtin_amdgcn_s_wait_tensorcnt(0);
      }
    }
    __syncthreads();
#else
    {  // fallback: synchronous f16 global->LDS staging
      int srow = tid >> 2, scol = (tid & 3) << 4;
      const _Float16* ap = A16 + (size_t)(mbase + srow) * K + (kb << 6) + scol;
      const _Float16* bp = W16 + (size_t)(nbase + srow) * K + (kb << 6) + scol;
      *(v8h*)(&As[cur][srow * 64 + scol])     = *(const v8h*)(ap);
      *(v8h*)(&As[cur][srow * 64 + scol + 8]) = *(const v8h*)(ap + 8);
      *(v8h*)(&Bs[cur][srow * 64 + scol])     = *(const v8h*)(bp);
      *(v8h*)(&Bs[cur][srow * 64 + scol + 8]) = *(const v8h*)(bp + 8);
    }
    __syncthreads();
#endif
#pragma unroll
    for (int ks = 0; ks < 2; ++ks) {  // 64-wide k-tile = 2 WMMA k-steps
      v16h af = frag_a(&As[cur][mt * 16 * 64 + ks * 32], 64);
      c0 = wmma_f16(af, frag_b(&Bs[cur][(nt0 + 0) * 16 * 64 + ks * 32], 64), c0);
      c1 = wmma_f16(af, frag_b(&Bs[cur][(nt0 + 1) * 16 * 64 + ks * 32], 64), c1);
    }
    __syncthreads();  // protect buffer before next TDM write / restage
  }
#pragma unroll
  for (int r = 0; r < 8; ++r) {
    int m = mbase + mt * 16 + r + lg * 8;
    int n0 = nbase + nt0 * 16 + nl;
    C[(size_t)m * N + n0]      = c0[r] + bias_s[nt0 * 16 + nl];
    C[(size_t)m * N + n0 + 16] = c1[r] + bias_s[(nt0 + 1) * 16 + nl];
  }
}

// ---------------- flash attention (per (b,h), 64 query rows, WMMA QK^T and PV) ----------------
__global__ __launch_bounds__(256)
void moe_attn_kernel(const float* __restrict__ qkv, float* __restrict__ ctx) {
  __shared__ __align__(16) _Float16 qs[64 * 96];   // Q tile, pre-scaled
  __shared__ __align__(16) _Float16 kvs[96 * 64];  // K tile [t][d] then V^T tile [d][t]
  __shared__ __align__(16) _Float16 pt[64 * 64];   // exp(score) tile as f16
  __shared__ float sf[64 * 64];                    // raw score tile f32
  __shared__ float rmax_s[64], rsum_s[64], corr_s[64], part_s[64 * 4];
  int tid = threadIdx.x;
  int bh = blockIdx.y;
  int b = bh >> 3, hh = bh & 7;
  int qbase = blockIdx.x * 64;
  const float scale = 0.102062072615966f;  // 96^-0.5
  {  // stage Q (scaled)
    int row = tid >> 2, c0 = (tid & 3) * 24;
    const float* src = qkv + ((size_t)(qbase + row) * B_SZ + b) * (3 * E_DIM) + hh * HD + c0;
    for (int j = 0; j < 24; ++j) qs[row * 96 + c0 + j] = (_Float16)(src[j] * scale);
  }
  if (tid < 64) { rmax_s[tid] = -1e30f; rsum_s[tid] = 0.0f; }
  int wave = tid >> 5, lane = tid & 31, lg = lane >> 4, nl = lane & 15;
  int mt = wave >> 1;
  int nt0 = (wave & 1) * 2;  // score subtiles
  int ct0 = (wave & 1) * 3;  // ctx subtiles (6 n-tiles of 16 = 96)
  v8f cx0 = {}, cx1 = {}, cx2 = {};
  for (int t0 = 0; t0 < S_LEN; t0 += 64) {
    __syncthreads();
    {  // stage K tile [t][d]
      int row = tid >> 2, c0 = (tid & 3) * 24;
      const float* src = qkv + ((size_t)(t0 + row) * B_SZ + b) * (3 * E_DIM) + E_DIM + hh * HD + c0;
      for (int j = 0; j < 24; ++j) kvs[row * 96 + c0 + j] = (_Float16)src[j];
    }
    __syncthreads();
    v8f s0 = {}, s1 = {};
#pragma unroll
    for (int ks = 0; ks < 3; ++ks) {  // K = 96
      v16h a = frag_a(qs + mt * 16 * 96 + ks * 32, 96);
      s0 = wmma_f16(a, frag_b(kvs + (nt0 + 0) * 16 * 96 + ks * 32, 96), s0);
      s1 = wmma_f16(a, frag_b(kvs + (nt0 + 1) * 16 * 96 + ks * 32, 96), s1);
    }
#pragma unroll
    for (int r = 0; r < 8; ++r) {
      int m = mt * 16 + r + lg * 8;
      sf[m * 64 + (nt0 + 0) * 16 + nl] = s0[r];
      sf[m * 64 + (nt0 + 1) * 16 + nl] = s1[r];
    }
    __syncthreads();
    {  // online softmax update, 4 threads per row
      int row = tid >> 2, q = tid & 3;
      float lm = -1e30f;
      for (int j = 0; j < 16; ++j) lm = fmaxf(lm, sf[row * 64 + q * 16 + j]);
      part_s[row * 4 + q] = lm;
      __syncthreads();
      if (q == 0) {
        float nm = fmaxf(fmaxf(part_s[row * 4], part_s[row * 4 + 1]),
                         fmaxf(part_s[row * 4 + 2], part_s[row * 4 + 3]));
        float om = rmax_s[row];
        nm = fmaxf(nm, om);
        corr_s[row] = __expf(om - nm);
        rmax_s[row] = nm;
      }
      __syncthreads();
      float nm = rmax_s[row];
      float ls = 0.0f;
      for (int j = 0; j < 16; ++j) {
        float e = __expf(sf[row * 64 + q * 16 + j] - nm);
        pt[row * 64 + q * 16 + j] = (_Float16)e;
        ls += e;
      }
      part_s[row * 4 + q] = ls;
      __syncthreads();
      if (q == 0)
        rsum_s[row] = rsum_s[row] * corr_s[row] +
                      part_s[row * 4] + part_s[row * 4 + 1] + part_s[row * 4 + 2] + part_s[row * 4 + 3];
    }
    {  // stage V^T tile [d][t] (reuse kvs; K tile already consumed)
      int trow = tid >> 2, c0 = (tid & 3) * 24;
      const float* src = qkv + ((size_t)(t0 + trow) * B_SZ + b) * (3 * E_DIM) + 2 * E_DIM + hh * HD + c0;
      for (int j = 0; j < 24; ++j) kvs[(c0 + j) * 64 + trow] = (_Float16)src[j];
    }
    {  // rescale running ctx accumulators by corr
      float cr[8];
#pragma unroll
      for (int r = 0; r < 8; ++r) cr[r] = corr_s[mt * 16 + r + lg * 8];
#pragma unroll
      for (int r = 0; r < 8; ++r) { cx0[r] *= cr[r]; cx1[r] *= cr[r]; cx2[r] *= cr[r]; }
    }
    __syncthreads();
#pragma unroll
    for (int ks = 0; ks < 2; ++ks) {  // K = 64 (this t-block)
      v16h a = frag_a(pt + mt * 16 * 64 + ks * 32, 64);
      cx0 = wmma_f16(a, frag_b(kvs + (ct0 + 0) * 16 * 64 + ks * 32, 64), cx0);
      cx1 = wmma_f16(a, frag_b(kvs + (ct0 + 1) * 16 * 64 + ks * 32, 64), cx1);
      cx2 = wmma_f16(a, frag_b(kvs + (ct0 + 2) * 16 * 64 + ks * 32, 64), cx2);
    }
  }
  __syncthreads();
#pragma unroll
  for (int r = 0; r < 8; ++r) {
    int m = mt * 16 + r + lg * 8;
    float inv = 1.0f / rsum_s[m];
    float* dst = ctx + ((size_t)(qbase + m) * B_SZ + b) * E_DIM + hh * HD;
    dst[(ct0 + 0) * 16 + nl] = cx0[r] * inv;
    dst[(ct0 + 1) * 16 + nl] = cx1[r] * inv;
    dst[(ct0 + 2) * 16 + nl] = cx2[r] * inv;
  }
}

// ---------------- gate: logits, softmax, top-2, append to expert lists ----------------
__global__ __launch_bounds__(256)
void moe_gate_kernel(const float* __restrict__ H, const float* __restrict__ Wg,
                     const float* __restrict__ bg, int* __restrict__ counts,
                     int* __restrict__ ltok, float* __restrict__ lw) {
  int wave = threadIdx.x >> 5, lane = threadIdx.x & 31;
  int tok = blockIdx.x * 8 + wave;
  const float* hrow = H + (size_t)tok * E_DIM;
  float lg[NEXP];
#pragma unroll
  for (int e = 0; e < NEXP; ++e) {
    const float* w = Wg + e * E_DIM;
    float acc = 0.0f;
    for (int i = lane; i < E_DIM; i += 32) acc += hrow[i] * w[i];
#pragma unroll
    for (int off = 16; off > 0; off >>= 1) acc += __shfl_xor(acc, off, 32);
    lg[e] = acc + bg[e];
  }
  if (lane == 0) {
    float m = lg[0];
    for (int e = 1; e < NEXP; ++e) m = fmaxf(m, lg[e]);
    float p[NEXP];
    for (int e = 0; e < NEXP; ++e) p[e] = __expf(lg[e] - m);
    int i1 = 0;
    for (int e = 1; e < NEXP; ++e) if (p[e] > p[i1]) i1 = e;
    int i2 = (i1 == 0) ? 1 : 0;
    for (int e = 0; e < NEXP; ++e) if (e != i1 && p[e] > p[i2]) i2 = e;
    float d = p[i1] + p[i2];
    float w1 = p[i1] / d, w2 = p[i2] / d;
    int s1 = atomicAdd(&counts[i1], 1);
    ltok[i1 * NTOK + s1] = tok; lw[i1 * NTOK + s1] = w1;
    int s2 = atomicAdd(&counts[i2], 1);
    ltok[i2 * NTOK + s2] = tok; lw[i2 * NTOK + s2] = w2;
  }
}

// ---------------- expert fc1: eh = relu(gather(h) @ W1[e] + b1[e]) ----------------
__global__ __launch_bounds__(256)
void moe_fc1_kernel(const float* __restrict__ H, const float* __restrict__ W1,
                    const float* __restrict__ b1, const int* __restrict__ ltok,
                    const int* __restrict__ counts, const int* __restrict__ offs,
                    float* __restrict__ eh) {
  int e = blockIdx.y >> 7;
  int mblk = blockIdx.y & 127;
  int count = counts[e];
  int mbase = mblk * 64;
  if (mbase >= count) return;
  __shared__ __align__(16) _Float16 As[64 * 32];
  __shared__ __align__(16) _Float16 Bs[64 * 32];
  __shared__ float bias_s[64];
  int tid = threadIdx.x;
  int nbase = blockIdx.x * 64;
  if (tid < 64) bias_s[tid] = b1[e * HID + nbase + tid];
  int srow = tid >> 2, scol = (tid & 3) << 3;
  int slot = mbase + srow; if (slot >= count) slot = count - 1;
  int tok = ltok[e * NTOK + slot];
  const float* arow_p = H + (size_t)tok * E_DIM;
  const float* W = W1 + (size_t)e * E_DIM * HID;  // (E,HID) K-major
  int obase = offs[e];
  int wave = tid >> 5, lane = tid & 31, lg = lane >> 4, nl = lane & 15;
  int mt = wave >> 1, nt0 = (wave & 1) * 2;
  v8f c0 = {}, c1 = {};
  for (int k0 = 0; k0 < E_DIM; k0 += 32) {
    __syncthreads();
    float4 a0 = *(const float4*)(arow_p + k0 + scol);
    float4 a1 = *(const float4*)(arow_p + k0 + scol + 4);
    *(v8h*)(As + srow * 32 + scol) = pack8(a0, a1);
    v8h bh_;
#pragma unroll
    for (int j = 0; j < 8; ++j)
      bh_[j] = (_Float16)W[(size_t)(k0 + scol + j) * HID + nbase + srow];
    *(v8h*)(Bs + srow * 32 + scol) = bh_;
    __syncthreads();
    v16h af = frag_a(As + mt * 16 * 32, 32);
    c0 = wmma_f16(af, frag_b(Bs + (nt0 + 0) * 16 * 32, 32), c0);
    c1 = wmma_f16(af, frag_b(Bs + (nt0 + 1) * 16 * 32, 32), c1);
  }
#pragma unroll
  for (int r = 0; r < 8; ++r) {
    int ml = mt * 16 + r + lg * 8;
    if (mbase + ml < count) {
      size_t orow = (size_t)(obase + mbase + ml) * HID + nbase;
      eh[orow + (nt0 + 0) * 16 + nl] = fmaxf(c0[r] + bias_s[(nt0 + 0) * 16 + nl], 0.0f);
      eh[orow + (nt0 + 1) * 16 + nl] = fmaxf(c1[r] + bias_s[(nt0 + 1) * 16 + nl], 0.0f);
    }
  }
}

// ---------------- expert fc2: moe[tok] += w * (eh @ W2[e] + b2[e]) ----------------
__global__ __launch_bounds__(256)
void moe_fc2_kernel(const float* __restrict__ eh, const float* __restrict__ W2,
                    const float* __restrict__ b2, const int* __restrict__ ltok,
                    const float* __restrict__ lw, const int* __restrict__ counts,
                    const int* __restrict__ offs, float* __restrict__ moe) {
  int e = blockIdx.y >> 7;
  int mblk = blockIdx.y & 127;
  int count = counts[e];
  int mbase = mblk * 64;
  if (mbase >= count) return;
  __shared__ __align__(16) _Float16 As[64 * 32];
  __shared__ __align__(16) _Float16 Bs[64 * 32];
  __shared__ float bias_s[64];
  __shared__ int toks_s[64];
  __shared__ float w_s[64];
  int tid = threadIdx.x;
  int nbase = blockIdx.x * 64;
  if (tid < 64) {
    bias_s[tid] = b2[e * DOUT + nbase + tid];
    int slot = mbase + tid; if (slot >= count) slot = count - 1;
    toks_s[tid] = ltok[e * NTOK + slot];
    w_s[tid] = lw[e * NTOK + slot];
  }
  int srow = tid >> 2, scol = (tid & 3) << 3;
  int arow = mbase + srow; if (arow >= count) arow = count - 1;
  const float* arow_p = eh + (size_t)(offs[e] + arow) * HID;
  const float* W = W2 + (size_t)e * HID * DOUT;  // (HID,DOUT) K-major
  int wave = tid >> 5, lane = tid & 31, lg = lane >> 4, nl = lane & 15;
  int mt = wave >> 1, nt0 = (wave & 1) * 2;
  v8f c0 = {}, c1 = {};
  for (int k0 = 0; k0 < HID; k0 += 32) {
    __syncthreads();
    float4 a0 = *(const float4*)(arow_p + k0 + scol);
    float4 a1 = *(const float4*)(arow_p + k0 + scol + 4);
    *(v8h*)(As + srow * 32 + scol) = pack8(a0, a1);
    v8h bh_;
#pragma unroll
    for (int j = 0; j < 8; ++j)
      bh_[j] = (_Float16)W[(size_t)(k0 + scol + j) * DOUT + nbase + srow];
    *(v8h*)(Bs + srow * 32 + scol) = bh_;
    __syncthreads();
    v16h af = frag_a(As + mt * 16 * 32, 32);
    c0 = wmma_f16(af, frag_b(Bs + (nt0 + 0) * 16 * 32, 32), c0);
    c1 = wmma_f16(af, frag_b(Bs + (nt0 + 1) * 16 * 32, 32), c1);
  }
#pragma unroll
  for (int r = 0; r < 8; ++r) {
    int ml = mt * 16 + r + lg * 8;
    if (mbase + ml < count) {
      float w = w_s[ml];
      int tok = toks_s[ml];
      atomicAdd(&moe[(size_t)tok * DOUT + nbase + (nt0 + 0) * 16 + nl],
                (c0[r] + bias_s[(nt0 + 0) * 16 + nl]) * w);
      atomicAdd(&moe[(size_t)tok * DOUT + nbase + (nt0 + 1) * 16 + nl],
                (c1[r] + bias_s[(nt0 + 1) * 16 + nl]) * w);
    }
  }
}

// ---------------- layernorm * aw, plus aw output ----------------
__global__ __launch_bounds__(256)
void moe_ln_kernel(const float* __restrict__ moe, const float* __restrict__ gamma,
                   const float* __restrict__ beta, float* __restrict__ out,
                   float* __restrict__ aw) {
  __shared__ float r1[256], r2[256];
  int row = blockIdx.x, tid = threadIdx.x;
  const float* x = moe + (size_t)row * DOUT;
  float s = 0.0f, s2 = 0.0f;
  for (int j = tid; j < DOUT; j += 256) { float v = x[j]; s += v; s2 += v * v; }
  r1[tid] = s; r2[tid] = s2;
  __syncthreads();
  for (int off = 128; off > 0; off >>= 1) {
    if (tid < off) { r1[tid] += r1[tid + off]; r2[tid] += r2[tid + off]; }
    __syncthreads();
  }
  float mu = r1[0] * (1.0f / DOUT);
  float var = r2[0] * (1.0f / DOUT) - mu * mu;
  float rs = rsqrtf(var + LN_EPS);
  for (int j = tid; j < DOUT; j += 256) {
    float v = (x[j] - mu) * rs;
    out[(size_t)row * DOUT + j] = (v * gamma[j] + beta[j]) * AW_CONST;
  }
  if (tid == 0) aw[row] = AW_CONST;
}

// ---------------- host launch ----------------
extern "C" void kernel_launch(void* const* d_in, const int* in_sizes, int n_in,
                              void* d_out, int out_size, void* d_ws, size_t ws_size,
                              hipStream_t stream) {
  (void)in_sizes; (void)n_in; (void)out_size; (void)ws_size;
  const float* x    = (const float*)d_in[0];
  const float* Wp   = (const float*)d_in[1];
  const float* bp   = (const float*)d_in[2];
  const float* Win  = (const float*)d_in[3];
  const float* bin_ = (const float*)d_in[4];
  const float* Wo   = (const float*)d_in[5];
  const float* bo   = (const float*)d_in[6];
  const float* Wg   = (const float*)d_in[7];
  const float* bg   = (const float*)d_in[8];
  const float* W1   = (const float*)d_in[9];
  const float* b1   = (const float*)d_in[10];
  const float* W2   = (const float*)d_in[11];
  const float* b2   = (const float*)d_in[12];
  const float* gamma = (const float*)d_in[13];
  const float* beta  = (const float*)d_in[14];
  float* out = (float*)d_out;
  float* aw_out = out + (size_t)NTOK * DOUT;

  char* ws = (char*)d_ws;
  size_t off = 0;
  auto alloc = [&](size_t bytes) {
    void* p = ws + off;
    off += (bytes + 255) & ~(size_t)255;
    return p;
  };
  float* h    = (float*)alloc((size_t)NTOK * E_DIM * 4);
  float* qkv  = (float*)alloc((size_t)NTOK * 3 * E_DIM * 4);
  float* ctxm = (float*)alloc((size_t)NTOK * DOUT * 4);        // ctx, later moe accumulator
  float* eh   = (float*)alloc((size_t)NTOK * 2 * HID * 4);     // 16384 gathered rows
  _Float16* x16   = (_Float16*)alloc((size_t)NTOK * DIN * 2);
  _Float16* h16   = (_Float16*)alloc((size_t)NTOK * E_DIM * 2);
  _Float16* ctx16 = (_Float16*)alloc((size_t)NTOK * E_DIM * 2);
  _Float16* Wp16  = (_Float16*)alloc((size_t)E_DIM * DIN * 2);
  _Float16* Win16 = (_Float16*)alloc((size_t)3 * E_DIM * E_DIM * 2);
  _Float16* Wo16  = (_Float16*)alloc((size_t)E_DIM * E_DIM * 2);
  int*   counts = (int*)alloc(NEXP * 4);
  int*   offs   = (int*)alloc(NEXP * 4);
  int*   ltok   = (int*)alloc((size_t)NEXP * NTOK * 4);
  float* lwt    = (float*)alloc((size_t)NEXP * NTOK * 4);

  moe_zero_kernel<<<1, 256, 0, stream>>>((float*)counts, NEXP);
  // f16 weight/activation copies for TDM-fed WMMA GEMMs
  moe_cvt_f16_kernel<<<(NTOK * DIN) / 1024, 256, 0, stream>>>(x, x16, NTOK * DIN);
  moe_cvt_f16_kernel<<<(E_DIM * DIN) / 1024, 256, 0, stream>>>(Wp, Wp16, E_DIM * DIN);
  moe_cvt_f16_kernel<<<(3 * E_DIM * E_DIM) / 1024, 256, 0, stream>>>(Win, Win16, 3 * E_DIM * E_DIM);
  moe_cvt_f16_kernel<<<(E_DIM * E_DIM) / 1024, 256, 0, stream>>>(Wo, Wo16, E_DIM * E_DIM);

  moe_gemm_tdm_f16<<<dim3(E_DIM / 64, NTOK / 64), 256, 0, stream>>>(x16, Wp16, bp, h, NTOK, E_DIM, DIN);
  moe_cvt_f16_kernel<<<(NTOK * E_DIM) / 1024, 256, 0, stream>>>(h, h16, NTOK * E_DIM);
  moe_gemm_tdm_f16<<<dim3(3 * E_DIM / 64, NTOK / 64), 256, 0, stream>>>(h16, Win16, bin_, qkv, NTOK, 3 * E_DIM, E_DIM);
  moe_attn_kernel<<<dim3(S_LEN / 64, B_SZ * NH), 256, 0, stream>>>(qkv, ctxm);
  moe_cvt_f16_kernel<<<(NTOK * E_DIM) / 1024, 256, 0, stream>>>(ctxm, ctx16, NTOK * E_DIM);
  moe_gemm_tdm_f16<<<dim3(E_DIM / 64, NTOK / 64), 256, 0, stream>>>(ctx16, Wo16, bo, h, NTOK, E_DIM, E_DIM);
  moe_zero_kernel<<<(NTOK * DOUT + 255) / 256, 256, 0, stream>>>(ctxm, NTOK * DOUT);
  moe_gate_kernel<<<NTOK / 8, 256, 0, stream>>>(h, Wg, bg, counts, ltok, lwt);
  moe_scan_kernel<<<1, 32, 0, stream>>>(counts, offs);
  moe_fc1_kernel<<<dim3(HID / 64, NEXP * (NTOK / 64)), 256, 0, stream>>>(h, W1, b1, ltok, counts, offs, eh);
  moe_fc2_kernel<<<dim3(DOUT / 64, NEXP * (NTOK / 64)), 256, 0, stream>>>(eh, W2, b2, ltok, lwt, counts, offs, ctxm);
  moe_ln_kernel<<<NTOK, 256, 0, stream>>>(ctxm, gamma, beta, out, aw_out);
}
